// PwcNet_34462817583629
// MI455X (gfx1250) — matrix operand order repaired and checked by reading, not compiled, not run on hardware
//
#include <hip/hip_runtime.h>

// ---------------------------------------------------------------------------
// PWC-Net forward for gfx1250 (MI455X).
// Heavy 3x3 convs -> implicit GEMM on V_WMMA_F32_16X16X32_F16 (wave32).
//   M = 16 output pixels, N = 16 output channels, K = Cin*9 tiled by 32.
// Weight K-tiles staged in LDS via GLOBAL_LOAD_ASYNC_TO_LDS_B32 (ASYNCcnt)
// for full interior tiles; branchless sync staging for Cout/K tails.
// All gathers are branchless: clamped address + select-to-zero, so loads
// issue back-to-back with batched waits instead of per-element exec masking.
// f32->f16 packing via v_cvt_pkrtz_f16_f32.
// ---------------------------------------------------------------------------

typedef _Float16 v16h __attribute__((ext_vector_type(16)));
typedef _Float16 v2h  __attribute__((ext_vector_type(2)));
typedef __fp16   hf2  __attribute__((ext_vector_type(2)));
typedef float    v8f  __attribute__((ext_vector_type(8)));
typedef float    v4f  __attribute__((ext_vector_type(4)));

#define NEG_SLOPE 0.1f
#define BSTR 36  // LDS row stride (floats): 16B-aligned rows, conflict-free banks

#if defined(__has_builtin)
#  if __has_builtin(__builtin_amdgcn_global_load_async_to_lds_b32) && \
      __has_builtin(__builtin_amdgcn_s_wait_asynccnt)
#    define PWC_ASYNC 1
#  endif
#  if __has_builtin(__builtin_amdgcn_cvt_pkrtz)
#    define PWC_PKRTZ 1
#  endif
#endif
#ifndef PWC_ASYNC
#define PWC_ASYNC 0
#endif
#ifndef PWC_PKRTZ
#define PWC_PKRTZ 0
#endif

#define AS_GLOBAL __attribute__((address_space(1)))
#define AS_LDS    __attribute__((address_space(3)))

__device__ __forceinline__ v2h cvt2h(float a, float b) {
#if PWC_PKRTZ
  union { hf2 h; v2h v; } u;
  u.h = __builtin_amdgcn_cvt_pkrtz(a, b);
  return u.v;
#else
  v2h r; r[0] = (_Float16)a; r[1] = (_Float16)b; return r;
#endif
}

__device__ __forceinline__ int clampi(int v, int lo, int hi) {
  return v < lo ? lo : (v > hi ? hi : v);
}

// ---------------------------------------------------------------------------
// Implicit-GEMM 3x3 conv (stride/pad/dilation), WMMA f16->f32.
// Input  read from channel window [icoff, icoff+Cin) of a buffer with ictot ch.
// Output written to channel window [ocoff, ocoff+Cout) of buffer with octot ch.
// ---------------------------------------------------------------------------
__global__ __launch_bounds__(256) void pwc_conv3x3_wmma(
    const float* __restrict__ in, const float* __restrict__ wgt,
    const float* __restrict__ bias, float* __restrict__ out,
    int Bn, int Cin, int H, int W, int icoff, int ictot,
    int Cout, int Hout, int Wout, int ocoff, int octot,
    int stride, int pad, int dil, int relu)
{
  __shared__ __align__(16) float sB[16 * BSTR];  // weight tile, [n][k], f32

  const int lane  = threadIdx.x & 31;
  const int wave  = threadIdx.x >> 5;
  const int khalf = lane >> 4;       // which K half this lane carries
  const int mrow  = lane & 15;       // A row (pixel) / B-D column (cout)
  const int cBase = blockIdx.x * 16;
  const long HWo  = (long)Hout * Wout;
  const long Npix = (long)Bn * HWo;
  const long pixBase = ((long)blockIdx.y * 8 + wave) * 16;
  const int Ktot = Cin * 9;
  const long HWi = (long)H * W;

  // Decode this lane's pixel (rows of A).
  long pix = pixBase + mrow;
  const bool pvalid = pix < Npix;
  long pc  = pvalid ? pix : 0;
  int bb   = (int)(pc / HWo);
  int rem  = (int)(pc - (long)bb * HWo);
  int oh   = rem / Wout, ow = rem - oh * Wout;
  int ihb  = oh * stride - pad, iwb = ow * stride - pad;
  const float* inb = in + ((long)bb * ictot + icoff) * HWi;

  const bool fullN = (cBase + 16) <= Cout;

  v8f acc = {0.f, 0.f, 0.f, 0.f, 0.f, 0.f, 0.f, 0.f};

  for (int kk = 0; kk < Ktot; kk += 32) {
#if PWC_ASYNC
    if (fullN && (kk + 32) <= Ktot) {
      // Async DMA of the full 16x32 f32 weight tile into LDS (ASYNCcnt path).
      const int t0 = threadIdx.x;
      const int n0 = t0 >> 5, k0 = t0 & 31;
      const int t1 = t0 + 256;
      const int n1 = t1 >> 5, k1 = t1 & 31;
      const float* g0 = wgt + (long)(cBase + n0) * Ktot + kk + k0;
      const float* g1 = wgt + (long)(cBase + n1) * Ktot + kk + k1;
      __builtin_amdgcn_global_load_async_to_lds_b32(
          (AS_GLOBAL int*)g0, (AS_LDS int*)&sB[n0 * BSTR + k0], 0, 0);
      __builtin_amdgcn_global_load_async_to_lds_b32(
          (AS_GLOBAL int*)g1, (AS_LDS int*)&sB[n1 * BSTR + k1], 0, 0);
      __builtin_amdgcn_s_wait_asynccnt(0);
    } else
#endif
    {
      // Branchless sync staging (handles Cout / K tails with zero fill).
#pragma unroll
      for (int h = 0; h < 2; ++h) {
        int t  = threadIdx.x + h * 256;
        int n  = t >> 5, kq = t & 31;
        int k  = kk + kq, co = cBase + n;
        int kc = k < Ktot ? k : (Ktot - 1);
        int cc = co < Cout ? co : (Cout - 1);
        float v = wgt[(long)cc * Ktot + kc];
        sB[n * BSTR + kq] = ((k < Ktot) && (co < Cout)) ? v : 0.f;
      }
    }
    __syncthreads();

    if (kk + 32 < Ktot)
      __builtin_prefetch(&wgt[(long)cBase * Ktot + kk + 32], 0, 1);

    // A fragment: 16x32 f16, contiguous k order: lane half h holds k = 16h + e.
    // Branchless gather: clamp address, always load, select to zero.
    union { v16h v; v2h p[8]; } A;
#pragma unroll
    for (int e2 = 0; e2 < 8; ++e2) {
      float f[2];
#pragma unroll
      for (int q = 0; q < 2; ++q) {
        int e = 2 * e2 + q;
        int kloc = khalf * 16 + e;
        int k  = kk + kloc;
        int kc = k < Ktot ? k : 0;
        int ci = kc / 9; int r = kc - ci * 9;
        int kh = r / 3, kw = r - kh * 3;
        int ih = ihb + kh * dil, iw = iwb + kw * dil;
        bool ok = pvalid && (k < Ktot) &&
                  ((unsigned)ih < (unsigned)H) && ((unsigned)iw < (unsigned)W);
        long off = ok ? (ci * HWi + (long)ih * W + iw) : 0;
        float v = inb[off];
        f[q] = ok ? v : 0.f;
      }
      A.p[e2] = cvt2h(f[0], f[1]);
    }

    // B fragment: lanes 0-15 carry K=0..15, lanes 16-31 K=16..31 of column mrow.
    // Contiguous 64B per lane -> ds_load_b128 x4, then packed f32->f16 converts.
    const float* bp = &sB[mrow * BSTR + khalf * 16];
    v4f w0 = *(const v4f*)(bp + 0);
    v4f w1 = *(const v4f*)(bp + 4);
    v4f w2 = *(const v4f*)(bp + 8);
    v4f w3 = *(const v4f*)(bp + 12);
    union { v16h v; v2h p[8]; } Bf;
    Bf.p[0] = cvt2h(w0[0], w0[1]);
    Bf.p[1] = cvt2h(w0[2], w0[3]);
    Bf.p[2] = cvt2h(w1[0], w1[1]);
    Bf.p[3] = cvt2h(w1[2], w1[3]);
    Bf.p[4] = cvt2h(w2[0], w2[1]);
    Bf.p[5] = cvt2h(w2[2], w2[3]);
    Bf.p[6] = cvt2h(w3[0], w3[1]);
    Bf.p[7] = cvt2h(w3[2], w3[3]);

    acc = __builtin_amdgcn_wmma_f32_16x16x32_f16(false, A.v, false, Bf.v,
                                                 (short)0, acc, false, false);
    __syncthreads();
  }

  // Epilogue: D 16x16 f32 layout -> lane holds column n=lane&15, rows khalf*8+r.
  const int co = cBase + mrow;
  if (co < Cout) {
    const float bv = bias[co];
#pragma unroll
    for (int r = 0; r < 8; ++r) {
      long m = pixBase + khalf * 8 + r;
      if (m < Npix) {
        float v = acc[r] + bv;
        if (relu) v = (v >= 0.f) ? v : NEG_SLOPE * v;
        int b2  = (int)(m / HWo);
        int rm  = (int)(m - (long)b2 * HWo);
        int oh2 = rm / Wout, ow2 = rm - oh2 * Wout;
        out[(((long)b2 * octot + ocoff + co) * Hout + oh2) * Wout + ow2] = v;
      }
    }
  }
}

// ---------------------------------------------------------------------------
// Cost volume with fused leaky-ReLU: out[b, (dy+4)*9+(dx+4), y, x]
// ---------------------------------------------------------------------------
__global__ void pwc_corr(const float* __restrict__ f1, const float* __restrict__ f2,
                         float* __restrict__ out, int Bn, int C, int H, int W,
                         int ocoff, int octot)
{
  long idx = (long)blockIdx.x * blockDim.x + threadIdx.x;
  long tot = (long)Bn * 81 * H * W;
  if (idx >= tot) return;
  int x = (int)(idx % W); long t = idx / W;
  int y = (int)(t % H);   t /= H;
  int d = (int)(t % 81);  int bb = (int)(t / 81);
  int dy = d / 9 - 4, dx = d - (d / 9) * 9 - 4;
  int y2 = y + dy, x2 = x + dx;
  long HW = (long)H * W;
  float s = 0.f;
  if (y2 >= 0 && y2 < H && x2 >= 0 && x2 < W) {
    const float* p1 = f1 + (long)bb * C * HW + (long)y * W + x;
    const float* p2 = f2 + (long)bb * C * HW + (long)y2 * W + x2;
    for (int c = 0; c < C; ++c) s += p1[c * HW] * p2[c * HW];
  }
  s /= (float)C;
  s = (s >= 0.f) ? s : NEG_SLOPE * s;
  out[(((long)bb * octot + ocoff + d) * H + y) * W + x] = s;
}

// ---------------------------------------------------------------------------
// backwarp: bilinear sample with zeros padding + binarized validity mask.
// flow read from channels [foff, foff+2) of a buffer with ftot channels.
// ---------------------------------------------------------------------------
__global__ void pwc_backwarp(const float* __restrict__ f2,
                             const float* __restrict__ fb, int foff, int ftot,
                             float* __restrict__ out, int Bn, int C, int H, int W,
                             float flt)
{
  long idx = (long)blockIdx.x * blockDim.x + threadIdx.x;
  long tot = (long)Bn * H * W;
  if (idx >= tot) return;
  int x = (int)(idx % W);
  int y = (int)((idx / W) % H);
  int bb = (int)(idx / ((long)H * W));
  long HW = (long)H * W;

  float fx = fb[(((long)bb * ftot + foff + 0) * H + y) * W + x] * flt;
  float fy = fb[(((long)bb * ftot + foff + 1) * H + y) * W + x] * flt;
  float gx = -1.f + (2.f * x + 1.f) / (float)W + fx / ((W - 1.f) * 0.5f);
  float gy = -1.f + (2.f * y + 1.f) / (float)H + fy / ((H - 1.f) * 0.5f);
  float sx = (gx + 1.f) * (0.5f * W) - 0.5f;
  float sy = (gy + 1.f) * (0.5f * H) - 0.5f;
  float x0f = floorf(sx), y0f = floorf(sy);
  float wx1 = sx - x0f, wy1 = sy - y0f;
  int x0 = (int)x0f, y0 = (int)y0f, x1 = x0 + 1, y1 = y0 + 1;
  float vx0 = (x0 >= 0 && x0 < W) ? 1.f : 0.f;
  float vx1 = (x1 >= 0 && x1 < W) ? 1.f : 0.f;
  float vy0 = (y0 >= 0 && y0 < H) ? 1.f : 0.f;
  float vy1 = (y1 >= 0 && y1 < H) ? 1.f : 0.f;
  float w00 = (1.f - wx1) * (1.f - wy1) * vx0 * vy0;
  float w01 = wx1 * (1.f - wy1) * vx1 * vy0;
  float w10 = (1.f - wx1) * wy1 * vx0 * vy1;
  float w11 = wx1 * wy1 * vx1 * vy1;
  float m = ((w00 + w01 + w10 + w11) > 0.999f) ? 1.f : 0.f;
  int x0c = clampi(x0, 0, W - 1), x1c = clampi(x1, 0, W - 1);
  int y0c = clampi(y0, 0, H - 1), y1c = clampi(y1, 0, H - 1);
  long i00 = (long)y0c * W + x0c, i01 = (long)y0c * W + x1c;
  long i10 = (long)y1c * W + x0c, i11 = (long)y1c * W + x1c;

  for (int c = 0; c < C; ++c) {
    const float* p = f2 + ((long)bb * C + c) * HW;
    float v = w00 * p[i00] + w01 * p[i01] + w10 * p[i10] + w11 * p[i11];
    out[((long)bb * C + c) * HW + (long)y * W + x] = v * m;
  }
}

// ---------------------------------------------------------------------------
// ConvTranspose2d k=4, s=2, p=1, Cout = 2; torch weight layout (Cin, 2, 4, 4).
// ---------------------------------------------------------------------------
__global__ void pwc_deconv4(const float* __restrict__ in, const float* __restrict__ w,
                            const float* __restrict__ bias, float* __restrict__ out,
                            int Bn, int Cin, int Hin, int Win, int ocoff, int octot)
{
  int Ho = Hin * 2, Wo = Win * 2;
  long tot = (long)Bn * 2 * Ho * Wo;
  long idx = (long)blockIdx.x * blockDim.x + threadIdx.x;
  if (idx >= tot) return;
  int ox = (int)(idx % Wo); long t = idx / Wo;
  int oy = (int)(t % Ho);   t /= Ho;
  int co = (int)(t % 2);    int bb = (int)(t / 2);
  long HWi = (long)Hin * Win;
  float s = bias[co];
  for (int ky = 0; ky < 4; ++ky) {
    int ty = oy + 1 - ky;
    if (ty & 1) continue;
    int iy = ty >> 1;
    if (iy < 0 || iy >= Hin) continue;
    for (int kx = 0; kx < 4; ++kx) {
      int tx = ox + 1 - kx;
      if (tx & 1) continue;
      int ix = tx >> 1;
      if (ix < 0 || ix >= Win) continue;
      const float* ip = in + (long)bb * Cin * HWi + (long)iy * Win + ix;
      const float* wp = w + (long)co * 16 + ky * 4 + kx;
      float a = 0.f;
      for (int ci = 0; ci < Cin; ++ci)
        a += ip[(long)ci * HWi] * wp[(long)ci * 32];
      s += a;
    }
  }
  out[(((long)bb * octot + ocoff + co) * Ho + oy) * Wo + ox] = s;
}

__global__ void pwc_copy_ch(const float* __restrict__ src, float* __restrict__ dst,
                            int Bn, int C, long HW, int ocoff, int octot)
{
  long idx = (long)blockIdx.x * blockDim.x + threadIdx.x;
  long tot = (long)Bn * C * HW;
  if (idx >= tot) return;
  long p = idx % HW; long t = idx / HW;
  int c = (int)(t % C); int bb = (int)(t / C);
  dst[((long)bb * octot + ocoff + c) * HW + p] = src[idx];
}

__global__ void pwc_add(const float* __restrict__ a, const float* __restrict__ b,
                        float* __restrict__ o, long n)
{
  long idx = (long)blockIdx.x * blockDim.x + threadIdx.x;
  if (idx < n) o[idx] = a[idx] + b[idx];
}

// ---------------------------------------------------------------------------
// Host orchestration
// ---------------------------------------------------------------------------
namespace {
struct Arena {
  char* base; size_t off;
  float* alloc(size_t elems) {
    size_t bytes = (elems * sizeof(float) + 255) & ~(size_t)255;
    float* p = (float*)(base + off);
    off += bytes;
    return p;
  }
};

// pytree (sorted-key) slot of each decoder conv inside a level:
// sorted keys: fiv(0) fou(1) one(2) six(3) thr(4) two(5) upfeat(6) upflow(7)
const int DEC_SLOT[8] = {2, 5, 4, 1, 0, 3, 6, 7}; // one,two,thr,fou,fiv,six,upfeat,upflow
const int EC[7] = {1, 16, 32, 64, 96, 128, 196};
const int EH[7] = {512, 256, 128, 64, 32, 16, 8};
const int RC[8] = {565, 128, 128, 128, 96, 64, 32, 2};
const int RD[7] = {1, 2, 4, 8, 16, 1, 1};
}

extern "C" void kernel_launch(void* const* d_in, const int* in_sizes, int n_in,
                              void* d_out, int out_size, void* d_ws, size_t ws_size,
                              hipStream_t stream) {
  (void)in_sizes; (void)n_in; (void)out_size; (void)ws_size;
  const float* ten1 = (const float*)d_in[0];
  const float* ten2 = (const float*)d_in[1];
  auto P = [&](int i) { return (const float*)d_in[i]; };
  // params flatten order: dec('2'..'6') then ext then ref
  auto decBase = [](int L) { return (L == 6) ? 66 : 2 + (L - 2) * 16; };
  auto decW = [&](int L, int q) { return P(decBase(L) + DEC_SLOT[q] * 2); };
  auto decB = [&](int L, int q) { return P(decBase(L) + DEC_SLOT[q] * 2 + 1); };
  auto extW = [&](int l, int i) { return P(78 + l * 6 + i * 2); };
  auto extB = [&](int l, int i) { return P(79 + l * 6 + i * 2); };
  auto refW = [&](int i) { return P(114 + i * 2); };
  auto refB = [&](int i) { return P(115 + i * 2); };

  Arena ar{(char*)d_ws, 0};

  auto conv = [&](const float* in, int icoff, int ictot, int Cin, int H, int Wd,
                  const float* w, const float* b, float* out, int ocoff, int octot,
                  int Cout, int strd, int pd, int dl, int relu) {
    int Ho = (H + 2 * pd - (2 * dl + 1)) / strd + 1;
    int Wo = (Wd + 2 * pd - (2 * dl + 1)) / strd + 1;
    long Npix = 4L * Ho * Wo;
    dim3 grid((unsigned)((Cout + 15) / 16), (unsigned)((Npix + 127) / 128));
    pwc_conv3x3_wmma<<<grid, 256, 0, stream>>>(in, w, b, out, 4, Cin, H, Wd,
                                               icoff, ictot, Cout, Ho, Wo, ocoff,
                                               octot, strd, pd, dl, relu);
  };
  auto g1 = [](long n) { return dim3((unsigned)((n + 255) / 256)); };

  // ---- Feature pyramid (both images) ----
  float* t0 = ar.alloc(4L * 16 * 256 * 256);
  float* t1 = ar.alloc(4L * 16 * 256 * 256);
  float* feats[2][6];
  for (int im = 0; im < 2; ++im) {
    const float* x = im ? ten2 : ten1;
    int Ci = 1, Hi = 512;
    for (int l = 0; l < 6; ++l) {
      int Co = EC[l + 1], Ho = EH[l + 1];
      feats[im][l] = ar.alloc(4L * Co * Ho * Ho);
      conv(x, 0, Ci, Ci, Hi, Hi, extW(l, 0), extB(l, 0), t0, 0, Co, Co, 2, 1, 1, 1);
      conv(t0, 0, Co, Co, Ho, Ho, extW(l, 1), extB(l, 1), t1, 0, Co, Co, 1, 1, 1, 1);
      conv(t1, 0, Co, Co, Ho, Ho, extW(l, 2), extB(l, 2), feats[im][l], 0, Co, Co, 1, 1, 1, 1);
      x = feats[im][l]; Ci = Co; Hi = Ho;
    }
  }

  // ---- Decoder pyramid ----
  struct Lvl { int L, fidx, Cf, Hs; float flt; };
  const Lvl lv[5] = {{6, 5, 196, 8, 0.f},
                     {5, 4, 128, 16, 0.625f},
                     {4, 3, 96, 32, 1.25f},
                     {3, 2, 64, 64, 2.5f},
                     {2, 1, 32, 128, 5.f}};
  float* f2w = ar.alloc(4L * 32 * 128 * 128);  // largest warped f2 (L2)
  float* pFeat = nullptr; float* pFlow = nullptr; int pCfin = 0, pH = 0;

  for (int i = 0; i < 5; ++i) {
    const int L = lv[i].L, fi = lv[i].fidx, Cf = lv[i].Cf, Hs = lv[i].Hs;
    const int cur = (L == 6) ? 81 : 81 + Cf + 4;
    const int Cfin = cur + 448;
    float* fb = ar.alloc(4L * Cfin * Hs * Hs);
    float* fl = ar.alloc(4L * 2 * Hs * Hs);

    if (L == 6) {
      long tc = 4L * 81 * Hs * Hs;
      pwc_corr<<<g1(tc), 256, 0, stream>>>(feats[0][fi], feats[1][fi], fb,
                                           4, Cf, Hs, Hs, 448, Cfin);
    } else {
      const int offFlow = 448 + 81 + Cf, offUp = offFlow + 2;
      long td = 4L * 2 * Hs * Hs;
      pwc_deconv4<<<g1(td), 256, 0, stream>>>(pFlow, decW(L, 7), decB(L, 7), fb,
                                              4, 2, pH, pH, offFlow, Cfin);
      pwc_deconv4<<<g1(td), 256, 0, stream>>>(pFeat, decW(L, 6), decB(L, 6), fb,
                                              4, pCfin, pH, pH, offUp, Cfin);
      long tp = 4L * Hs * Hs;
      pwc_backwarp<<<g1(tp), 256, 0, stream>>>(feats[1][fi], fb, offFlow, Cfin,
                                               f2w, 4, Cf, Hs, Hs, lv[i].flt);
      long tc = 4L * 81 * Hs * Hs;
      pwc_corr<<<g1(tc), 256, 0, stream>>>(feats[0][fi], f2w, fb,
                                           4, Cf, Hs, Hs, 448, Cfin);
      long tcp = 4L * Cf * Hs * Hs;
      pwc_copy_ch<<<g1(tcp), 256, 0, stream>>>(feats[0][fi], fb, 4, Cf,
                                               (long)Hs * Hs, 448 + 81, Cfin);
    }

    // DenseNet stack, in place via disjoint channel windows (prepend new blocks)
    conv(fb, 448, Cfin, cur,       Hs, Hs, decW(L, 0), decB(L, 0), fb, 320, Cfin, 128, 1, 1, 1, 1);
    conv(fb, 320, Cfin, cur + 128, Hs, Hs, decW(L, 1), decB(L, 1), fb, 192, Cfin, 128, 1, 1, 1, 1);
    conv(fb, 192, Cfin, cur + 256, Hs, Hs, decW(L, 2), decB(L, 2), fb,  96, Cfin,  96, 1, 1, 1, 1);
    conv(fb,  96, Cfin, cur + 352, Hs, Hs, decW(L, 3), decB(L, 3), fb,  32, Cfin,  64, 1, 1, 1, 1);
    conv(fb,  32, Cfin, cur + 416, Hs, Hs, decW(L, 4), decB(L, 4), fb,   0, Cfin,  32, 1, 1, 1, 1);
    conv(fb,   0, Cfin, cur + 448, Hs, Hs, decW(L, 5), decB(L, 5), fl,   0, 2,      2, 1, 1, 1, 0);

    pFeat = fb; pFlow = fl; pCfin = Cfin; pH = Hs;
  }

  // ---- Refiner (dilated convs, leaky-ReLU on all but last) ----
  float* r0 = ar.alloc(4L * 128 * 128 * 128);
  float* r1 = ar.alloc(4L * 128 * 128 * 128);
  float* bufs[2] = {r0, r1};
  const float* rp = pFeat;
  for (int i = 0; i < 7; ++i) {
    float* dst = bufs[i & 1];
    conv(rp, 0, RC[i], RC[i], 128, 128, refW(i), refB(i), dst, 0, RC[i + 1],
         RC[i + 1], 1, RD[i], RD[i], (i < 6) ? 1 : 0);
    rp = dst;
  }

  // ---- flow + refinement -> d_out ----
  long n = 4L * 2 * 128 * 128;
  pwc_add<<<g1(n), 256, 0, stream>>>(pFlow, rp, (float*)d_out, n);
}
